// Model_81973745811463
// MI455X (gfx1250) — matrix operand order repaired and checked by reading, not compiled, not run on hardware
//
#include <hip/hip_runtime.h>
#include <hip/hip_bf16.h>

// ---------------------------------------------------------------------------
// Model: DGCNN-style point net. B=16, N=2048 points, f32 throughout.
// WMMA path: v_wmma_f32_16x16x32_bf16 for all feature GEMMs.
// Async path: global_load_async_to_lds_b32 stages A-tiles (full k-chunks).
// ---------------------------------------------------------------------------

#define NB   16
#define NPTS 2048
#define MTOT (NB * NPTS)   // 32768 rows
#define EPS  1e-5f

typedef __attribute__((ext_vector_type(16))) __bf16 v16bf;
typedef __attribute__((ext_vector_type(8)))  float  v8f;

// ---------------------------------------------------------------------------
// KNN: one wave per query row. Distances cached in LDS (8KB/row), then 64
// wave-parallel argmin passes (top-64 sorted ascending; top-16 is its prefix).
// ---------------------------------------------------------------------------
__global__ void knn_kernel(const float* __restrict__ pos, int* __restrict__ idxOut)
{
    __shared__ float dist[4][NPTS];               // 4 waves * 8KB = 32KB LDS
    const int wave = threadIdx.x >> 5;
    const int lane = threadIdx.x & 31;
    const int row  = blockIdx.x * 4 + wave;       // global point id
    const int b    = row >> 11;                   // / NPTS
    const int n    = row & (NPTS - 1);
    const float* pb = pos + (size_t)b * NPTS * 3;

    const float qx = pb[n * 3 + 0];
    const float qy = pb[n * 3 + 1];
    const float qz = pb[n * 3 + 2];

    for (int j = lane; j < NPTS; j += 32) {
        float dx = pb[j * 3 + 0] - qx;
        float dy = pb[j * 3 + 1] - qy;
        float dz = pb[j * 3 + 2] - qz;
        float d  = dx * dx + dy * dy + dz * dz;
        if (j == n) d = 1e10f;                    // mask diagonal
        dist[wave][j] = d;
    }
    __builtin_amdgcn_wave_barrier();

    int* out = idxOut + (size_t)row * 64;
    for (int it = 0; it < 64; ++it) {
        float best = 3.4e38f;
        int   bidx = 0x7fffffff;
        for (int j = lane; j < NPTS; j += 32) {
            float d = dist[wave][j];
            if (d < best) { best = d; bidx = j; }
        }
        // wave32 butterfly argmin (tie -> smaller index, matches sorted top_k)
        #pragma unroll
        for (int off = 16; off >= 1; off >>= 1) {
            float ob = __shfl_xor(best, off, 32);
            int   oi = __shfl_xor(bidx, off, 32);
            if (ob < best || (ob == best && oi < bidx)) { best = ob; bidx = oi; }
        }
        if (lane == 0) {
            out[it] = bidx;
            dist[wave][bidx] = 3.4e38f;           // remove winner
        }
        __builtin_amdgcn_wave_barrier();
    }
}

// ---------------------------------------------------------------------------
// Tiled GEMM + bias: H[M x FO] = A[M x K] @ W[K x FO] + bias.
// Block = 256 threads (8 waves) -> 64x32 output tile (4x2 WMMA subtiles).
// K-chunks of 32 staged in LDS (zero-padded), full chunks via async-to-LDS.
// Compile-time K/FO -> fully unrolled, branch-free fragment builds.
// ---------------------------------------------------------------------------
template<int K, int FO>
__global__ void gemm_bias_tiled(const float* __restrict__ A, int lda,
                                const float* __restrict__ W,
                                const float* __restrict__ bias,
                                float* __restrict__ H)
{
    constexpr int KCH = (K + 31) / 32;
    __shared__ float tA[64 * 32];                 // 8 KB  (rows x k)
    __shared__ float tW[32 * 32];                 // 4 KB  (k x cols)

    const int t  = threadIdx.x;                   // 0..255
    const int l  = t & 31;
    const int wv = t >> 5;                        // wave 0..7
    const int r  = l & 15;
    const int hi = l >> 4;                        // lane half
    const int tm = blockIdx.x * 64;
    const int tn = blockIdx.y * 32;
    const int mr = (wv >> 1) * 16;                // wave row sub-tile
    const int nc = (wv & 1) * 16;                 // wave col sub-tile

    // Low 32 bits of a flat shared pointer == LDS byte offset (ISA 10.2).
    const unsigned ldsA = (unsigned)(uintptr_t)&tA[0];

    v8f acc = {};
    for (int kc = 0; kc < KCH; ++kc) {
        const int k0 = kc * 32;

        // ---- stage A chunk: 64 rows x 32 k ----
        if (k0 + 32 <= K) {
            // full chunk: VGPR-bypassing async copy, 8 dwords/thread
            #pragma unroll
            for (int q = 0; q < 8; ++q) {
                int id  = q * 256 + t;            // 0..2047
                int row = id >> 5, kk = id & 31;
                const float* gp = A + (size_t)(tm + row) * lda + k0 + kk;
                unsigned la = ldsA + (unsigned)id * 4u;
                asm volatile("global_load_async_to_lds_b32 %0, %1, off"
                             :: "v"(la), "v"((unsigned long long)(uintptr_t)gp)
                             : "memory");
            }
            asm volatile("s_wait_asynccnt 0x0" ::: "memory");
        } else {
            // tail chunk: padded manual stores (select, not branch)
            #pragma unroll
            for (int q = 0; q < 8; ++q) {
                int id  = q * 256 + t;
                int row = id >> 5, kk = id & 31;
                tA[id] = (k0 + kk < K) ? A[(size_t)(tm + row) * lda + k0 + kk] : 0.f;
            }
        }
        // ---- stage W chunk: 32 k x 32 cols ----
        #pragma unroll
        for (int q = 0; q < 4; ++q) {
            int id = q * 256 + t;                 // 0..1023
            int kk = id >> 5, n = id & 31;
            tW[id] = (k0 + kk < K) ? W[(size_t)(k0 + kk) * FO + tn + n] : 0.f;
        }
        __syncthreads();

        // ---- fragments (ISA 7.12.2 16-bit layouts), all unconditional ----
        v16bf af, bf;
        #pragma unroll
        for (int v = 0; v < 8; ++v) {
            int ka = ((v < 4) ? 0 : 16) + hi * 8 + 2 * (v & 3);
            float2 a2 = *(const float2*)&tA[(mr + r) * 32 + ka];
            af[2 * v]     = (__bf16)a2.x;
            af[2 * v + 1] = (__bf16)a2.y;
            int kb = ((v < 4) ? 0 : 16) + hi * 8 + 2 * (v & 3);
            bf[2 * v]     = (__bf16)tW[kb * 32 + nc + r];
            bf[2 * v + 1] = (__bf16)tW[(kb + 1) * 32 + nc + r];
        }
        acc = __builtin_amdgcn_wmma_f32_16x16x32_bf16(
                  false, af, false, bf, (short)0, acc, false, false);
        __syncthreads();
    }
    // C/D layout: VGPR i -> row mr + i + hi*8, col nc + (lane&15)
    #pragma unroll
    for (int i = 0; i < 8; ++i) {
        int row = tm + mr + i + hi * 8;
        int col = tn + nc + r;
        H[(size_t)row * FO + col] = acc[i] + bias[col];
    }
}

// ---------------------------------------------------------------------------
// Neighbor gather + max over k neighbors. float4 (b128) loads; h is
// L2-resident (<=32MB vs 192MB L2) so gathers run at L2 bandwidth.
// ---------------------------------------------------------------------------
__global__ void gather_max_kernel(const float* __restrict__ H, int fo,
                                  const int* __restrict__ idx, int k,
                                  float* __restrict__ agg)
{
    const int c4 = fo >> 2;
    size_t tid = (size_t)blockIdx.x * blockDim.x + threadIdx.x;
    if (tid >= (size_t)MTOT * c4) return;
    const int i = (int)(tid / c4);
    const int c = (int)(tid % c4) * 4;
    const int bbase = (i >> 11) << 11;            // batch row base
    const int* ip = idx + (size_t)i * 64;

    float4 m = make_float4(-3.4e38f, -3.4e38f, -3.4e38f, -3.4e38f);
    for (int j = 0; j < k; ++j) {
        int nb = ip[j];
        const float4* p = (const float4*)(H + ((size_t)(bbase + nb) * fo + c));
        if (j + 1 < k)
            __builtin_prefetch(H + ((size_t)(bbase + ip[j + 1]) * fo + c), 0, 0);
        float4 v = *p;
        m.x = fmaxf(m.x, v.x); m.y = fmaxf(m.y, v.y);
        m.z = fmaxf(m.z, v.z); m.w = fmaxf(m.w, v.w);
    }
    *(float4*)(agg + (size_t)i * fo + c) = m;
}

// ---------------------------------------------------------------------------
// Per-column mean / biased variance over M rows (block-per-column reduce).
// ---------------------------------------------------------------------------
__global__ void col_stats_kernel(const float* __restrict__ X, int ld,
                                 float* __restrict__ mean, float* __restrict__ var)
{
    const int c = blockIdx.x;
    float s = 0.f, sq = 0.f;
    for (int r = threadIdx.x; r < MTOT; r += blockDim.x) {
        float v = X[(size_t)r * ld + c];
        s += v; sq += v * v;
    }
    __shared__ float ss[256], ssq[256];
    ss[threadIdx.x] = s; ssq[threadIdx.x] = sq;
    __syncthreads();
    for (int o = 128; o >= 1; o >>= 1) {
        if (threadIdx.x < o) {
            ss[threadIdx.x]  += ss[threadIdx.x + o];
            ssq[threadIdx.x] += ssq[threadIdx.x + o];
        }
        __syncthreads();
    }
    if (threadIdx.x == 0) {
        float m = ss[0] / (float)MTOT;
        mean[c] = m;
        var[c]  = ssq[0] / (float)MTOT - m * m;
    }
}

// ---------------------------------------------------------------------------
// BN + ReLU, writing into dst at a column offset (fused concat).
// ---------------------------------------------------------------------------
__global__ void bn_relu_store_kernel(const float* __restrict__ X, int ldx,
                                     const float* __restrict__ mean,
                                     const float* __restrict__ var,
                                     const float* __restrict__ gamma,
                                     const float* __restrict__ beta,
                                     float* __restrict__ dst, int ldd, int colOff, int C)
{
    size_t tid = (size_t)blockIdx.x * blockDim.x + threadIdx.x;
    if (tid >= (size_t)MTOT * C) return;
    const int r = (int)(tid / C);
    const int c = (int)(tid % C);
    float v = X[(size_t)r * ldx + c];
    float y = (v - mean[c]) * rsqrtf(var[c] + EPS) * gamma[c] + beta[c];
    dst[(size_t)r * ldd + colOff + c] = fmaxf(y, 0.f);
}

// Copy pos into x_all columns [0..2] and [163..165] (the two concat heads).
__global__ void copy_pos_kernel(const float* __restrict__ pos, float* __restrict__ x_all)
{
    size_t tid = (size_t)blockIdx.x * blockDim.x + threadIdx.x;
    if (tid >= (size_t)MTOT * 3) return;
    const int r = (int)(tid / 3);
    const int c = (int)(tid % 3);
    float v = pos[(size_t)r * 3 + c];
    x_all[(size_t)r * 454 + c]       = v;
    x_all[(size_t)r * 454 + 163 + c] = v;
}

// Per-batch max pool over N points: feat[MTOT x 256] -> pooled[16 x 256].
__global__ void pool_max_kernel(const float* __restrict__ feat, float* __restrict__ pooled)
{
    const int b = blockIdx.x;
    const int c = threadIdx.x;                    // 256 cols
    float m = -3.4e38f;
    const float* p = feat + ((size_t)b * NPTS) * 256 + c;
    for (int n = 0; n < NPTS; ++n) m = fmaxf(m, p[(size_t)n * 256]);
    pooled[b * 256 + c] = m;
}

// Head: out[16 x 40] = pooled @ W(256x40) + b. Tiny; plain dot products.
__global__ void final_out_kernel(const float* __restrict__ pooled,
                                 const float* __restrict__ W,
                                 const float* __restrict__ bias,
                                 float* __restrict__ out)
{
    int tid = blockIdx.x * blockDim.x + threadIdx.x;
    if (tid >= NB * 40) return;
    const int b = tid / 40, o = tid % 40;
    float s = bias[o];
    const float* p = pooled + b * 256;
    for (int c = 0; c < 256; ++c) s += p[c] * W[c * 40 + o];
    out[tid] = s;
}

// ---------------------------------------------------------------------------
// Orchestration
// ---------------------------------------------------------------------------
extern "C" void kernel_launch(void* const* d_in, const int* in_sizes, int n_in,
                              void* d_out, int out_size, void* d_ws, size_t ws_size,
                              hipStream_t stream)
{
    (void)in_sizes; (void)n_in; (void)out_size; (void)ws_size;
    const float* pos = (const float*)d_in[0];
    // d_in[1] = B (scalar), constant 16 here.
    struct P { const float *w, *b, *g, *be; };
    auto prm = [&](int base) {
        P p; p.w = (const float*)d_in[base]; p.b = (const float*)d_in[base + 1];
        p.g = (const float*)d_in[base + 2]; p.be = (const float*)d_in[base + 3];
        return p;
    };
    P g1a = prm(2), g1b = prm(6), g2a = prm(10), g2b = prm(14),
      d3a = prm(18), d3b = prm(22), fa = prm(26);
    const float* ow = (const float*)d_in[30];
    const float* ob = (const float*)d_in[31];

    char* ws = (char*)d_ws;
    int*   idx    = (int*)ws;                                   // 8 MB
    float* x_all  = (float*)(ws + (size_t)8   * 1024 * 1024);   // 32768 x 454 (59.5 MB)
    float* h      = (float*)(ws + (size_t)68  * 1024 * 1024);   // 32768 x 256 (32 MB)
    float* agg    = (float*)(ws + (size_t)100 * 1024 * 1024);   // 32768 x 128 (16 MB)
    float* x3     = (float*)(ws + (size_t)116 * 1024 * 1024);   // 32768 x 32  (4 MB)
    float* mean   = (float*)(ws + (size_t)120 * 1024 * 1024);
    float* var    = mean + 256;
    float* pooled = var + 256;                                  // 16 x 256

    const dim3 blk256(256);
    auto cdiv = [](size_t a, size_t b) { return (unsigned)((a + b - 1) / b); };

    // 1. KNN (single k=64 pass; k=16 indices are its prefix)
    knn_kernel<<<MTOT / 4, 128, 0, stream>>>(pos, idx);
    // 2. pos into both concat heads of x_all
    copy_pos_kernel<<<cdiv((size_t)MTOT * 3, 256), blk256, 0, stream>>>(pos, x_all);

    // GCN layer helper: gemm -> gather/max -> stats -> bn_relu into x_all
#define GCN_TAIL(PRM, FO, KNN_K, COLOFF)                                                 \
    do {                                                                                 \
        gather_max_kernel<<<cdiv((size_t)MTOT * ((FO) / 4), 256), blk256, 0, stream>>>(  \
            h, (FO), idx, (KNN_K), agg);                                                 \
        col_stats_kernel<<<(FO), blk256, 0, stream>>>(agg, (FO), mean, var);             \
        bn_relu_store_kernel<<<cdiv((size_t)MTOT * (FO), 256), blk256, 0, stream>>>(     \
            agg, (FO), mean, var, PRM.g, PRM.be, x_all, 454, (COLOFF), (FO));            \
    } while (0)

    gemm_bias_tiled<3, 32><<<dim3(MTOT / 64, 1), blk256, 0, stream>>>(pos, 3, g1a.w, g1a.b, h);
    GCN_TAIL(g1a, 32, 16, 3);                                   // x1 part a
    gemm_bias_tiled<35, 128><<<dim3(MTOT / 64, 4), blk256, 0, stream>>>(x_all, 454, g1b.w, g1b.b, h);
    GCN_TAIL(g1b, 128, 16, 35);                                 // x1 part b
    gemm_bias_tiled<3, 32><<<dim3(MTOT / 64, 1), blk256, 0, stream>>>(pos, 3, g2a.w, g2a.b, h);
    GCN_TAIL(g2a, 32, 64, 166);                                 // x2 part a
    gemm_bias_tiled<35, 128><<<dim3(MTOT / 64, 4), blk256, 0, stream>>>(x_all + 163, 454, g2b.w, g2b.b, h);
    GCN_TAIL(g2b, 128, 64, 198);                                // x2 part b
#undef GCN_TAIL

    // Dense branch: d3a (3->32) then d3b (32->128), BN directly on h.
    gemm_bias_tiled<3, 32><<<dim3(MTOT / 64, 1), blk256, 0, stream>>>(pos, 3, d3a.w, d3a.b, h);
    col_stats_kernel<<<32, blk256, 0, stream>>>(h, 32, mean, var);
    bn_relu_store_kernel<<<cdiv((size_t)MTOT * 32, 256), blk256, 0, stream>>>(
        h, 32, mean, var, d3a.g, d3a.be, x3, 32, 0, 32);
    gemm_bias_tiled<32, 128><<<dim3(MTOT / 64, 4), blk256, 0, stream>>>(x3, 32, d3b.w, d3b.b, h);
    col_stats_kernel<<<128, blk256, 0, stream>>>(h, 128, mean, var);
    bn_relu_store_kernel<<<cdiv((size_t)MTOT * 128, 256), blk256, 0, stream>>>(
        h, 128, mean, var, d3b.g, d3b.be, x_all, 454, 326, 128);

    // fa: 454 -> 256, BN+ReLU, then reuse x_all storage as feat (ld = 256).
    gemm_bias_tiled<454, 256><<<dim3(MTOT / 64, 8), blk256, 0, stream>>>(
        x_all, 454, fa.w, fa.b, h);
    col_stats_kernel<<<256, blk256, 0, stream>>>(h, 256, mean, var);
    bn_relu_store_kernel<<<cdiv((size_t)MTOT * 256, 256), blk256, 0, stream>>>(
        h, 256, mean, var, fa.g, fa.be, x_all, 256, 0, 256);

    // Per-batch max pool + head.
    pool_max_kernel<<<NB, blk256, 0, stream>>>(x_all, pooled);
    final_out_kernel<<<cdiv(NB * 40, 256), blk256, 0, stream>>>(pooled, ow, ob, (float*)d_out);
}
